// GNNEncoder_3573412790413
// MI455X (gfx1250) — matrix operand order repaired and checked by reading, not compiled, not run on hardware
//
#include <hip/hip_runtime.h>
#include <hip/hip_bf16.h>
#include <math.h>

#define NNODES 50000
#define NEDGES 800000
#define DIM 256
#define LN_EPS 1e-5f

typedef __attribute__((ext_vector_type(16))) __bf16 v16bf;
typedef __attribute__((ext_vector_type(8)))  float  v8f;
typedef __attribute__((ext_vector_type(4)))  unsigned int u32x4;

union FragBF {
    v16bf v;
    u32x4 q[2];
};

// ---------------------------------------------------------------------------
// Pack fp32 W (256x256 row-major, W[k][n]) into bf16 WMMA B-fragment order so
// the GEMM inner loop loads each B fragment as 2 contiguous b128 loads.
// Fragment layout used here (B = 32x16, contiguous K, same order as the A loads):
//   lane l holds column n = l&15; halfs j=0..15 hold K = ((l>>4)*16 + j).
// packed[((ktile*16 + ntile)*32 + lane)*16 + j]
// ---------------------------------------------------------------------------
__global__ void k_pack_w(const float* __restrict__ W, __bf16* __restrict__ out) {
    int idx   = blockIdx.x * 256 + threadIdx.x;   // 65536 elements total
    int j     = idx & 15;
    int lane  = (idx >> 4) & 31;
    int tile  = idx >> 9;                         // 0..127
    int ntile = tile & 15;
    int ktile = tile >> 4;
    int k = ktile * 32 + ((lane >> 4) << 4) + j;
    int n = (ntile << 4) + (lane & 15);
    out[idx] = (__bf16)W[k * 256 + n];
}

// ---------------------------------------------------------------------------
// Edge-parallel SpMM: agg[dst] += w * x[src].  One wave32 per edge, 8 floats
// per lane. agg (51MB) is L2-resident, so global_atomic_add_f32 RMWs stay in
// the 192MB L2 on MI455X.
// ---------------------------------------------------------------------------
__global__ void k_spmm(const int* __restrict__ src, const int* __restrict__ dst,
                       const float* __restrict__ ew, const float* __restrict__ x,
                       float* __restrict__ agg) {
    int wid  = (blockIdx.x * blockDim.x + threadIdx.x) >> 5;   // edge id
    int lane = threadIdx.x & 31;
    int s = src[wid];
    int d = dst[wid];
    float w = ew[wid];
    const float4* xs = (const float4*)(x + (size_t)s * DIM);
    float4 v0 = xs[lane];        // cols [4*lane, 4*lane+3]
    float4 v1 = xs[lane + 32];   // cols [128+4*lane, ...]
    float* ad = agg + (size_t)d * DIM;
    int c0 = lane * 4;
    unsafeAtomicAdd(ad + c0 + 0,       w * v0.x);
    unsafeAtomicAdd(ad + c0 + 1,       w * v0.y);
    unsafeAtomicAdd(ad + c0 + 2,       w * v0.z);
    unsafeAtomicAdd(ad + c0 + 3,       w * v0.w);
    unsafeAtomicAdd(ad + 128 + c0 + 0, w * v1.x);
    unsafeAtomicAdd(ad + 128 + c0 + 1, w * v1.y);
    unsafeAtomicAdd(ad + 128 + c0 + 2, w * v1.z);
    unsafeAtomicAdd(ad + 128 + c0 + 3, w * v1.w);
}

__device__ __forceinline__ float gelu_exact(float v) {
    return 0.5f * v * (1.0f + erff(v * 0.70710678118654752f));
}

// ---------------------------------------------------------------------------
// Fused  OUT = epilogue( (X [+ AGG]) @ W + bias )
// Block = 16 rows x 256 cols (LayerNorm row is block-local). 256 threads =
// 8 waves; wave w computes column tiles 2w, 2w+1. K = 8 steps of 32 (bf16).
// A fragment (16x32 bf16): lane l holds row m=l&15; halfs 0..15 = K kofs..kofs+15
// with kofs = (l<16 ? 0 : 16) -> two b128 LDS
// loads per step.
// ---------------------------------------------------------------------------
template<bool ADD_AGG, bool LN_GELU>
__global__ void k_gemm_fused(const float* __restrict__ X, const float* __restrict__ AGG,
                             const __bf16* __restrict__ Wp, const float* __restrict__ bias,
                             const float* __restrict__ gamma, const float* __restrict__ beta,
                             float* __restrict__ OUT) {
    __shared__ alignas(16) __bf16 As[16 * 256];  // 8 KB  A tile (bf16)
    __shared__ float Hs[16 * 256];               // 16 KB pre-LN activations
    __shared__ float RS[16][16];
    __shared__ float RSS[16][16];

    const int tid  = threadIdx.x;
    const int lane = tid & 31;
    const int wave = tid >> 5;
    const size_t rowbase = (size_t)blockIdx.x * 16;   // 50000 = 16*3125 exactly

    // ---- Stage A tile: bf16(X [+ AGG]) into LDS (16 fp32 per thread) ----
    {
        const int r  = tid >> 4;
        const int cs = (tid & 15) << 4;
        const float4* xr = (const float4*)(X + (rowbase + r) * DIM + cs);
        const float4* ar = ADD_AGG ? (const float4*)(AGG + (rowbase + r) * DIM + cs) : nullptr;
        __bf16* a = As + r * 256 + cs;
        #pragma unroll
        for (int q = 0; q < 4; ++q) {
            float4 xv = xr[q];
            if (ADD_AGG) {
                float4 av = ar[q];
                xv.x += av.x; xv.y += av.y; xv.z += av.z; xv.w += av.w;
            }
            a[q * 4 + 0] = (__bf16)xv.x;
            a[q * 4 + 1] = (__bf16)xv.y;
            a[q * 4 + 2] = (__bf16)xv.z;
            a[q * 4 + 3] = (__bf16)xv.w;
        }
    }
    __syncthreads();

    // ---- WMMA main loop: each wave owns ntiles {2w, 2w+1} ----
    v8f c0 = {};
    v8f c1 = {};
    const int arow  = lane & 15;
    const int kaofs = (lane < 16) ? 0 : 2;       // u32x4 index of low K chunk
    const u32x4* wq = (const u32x4*)Wp;
    #pragma unroll
    for (int kk = 0; kk < 8; ++kk) {
        FragBF a, b0, b1;
        const u32x4* ap = (const u32x4*)(As + arow * 256 + kk * 32);
        a.q[0] = ap[kaofs];          // K kofs..kofs+7
        a.q[1] = ap[kaofs + 1];      // K kofs+8..kofs+15
        const u32x4* bp0 = wq + ((size_t)((kk * 16 + 2 * wave) * 32 + lane) << 1);
        b0.q[0] = bp0[0];
        b0.q[1] = bp0[1];
        const u32x4* bp1 = wq + ((size_t)((kk * 16 + 2 * wave + 1) * 32 + lane) << 1);
        b1.q[0] = bp1[0];
        b1.q[1] = bp1[1];
        c0 = __builtin_amdgcn_wmma_f32_16x16x32_bf16(false, a.v, false, b0.v, (short)0, c0, false, false);
        c1 = __builtin_amdgcn_wmma_f32_16x16x32_bf16(false, a.v, false, b1.v, (short)0, c1, false, false);
    }

    // ---- Scatter accumulators (+bias) into Hs (C/D layout: vgpr r -> row r+8*(l>>4), col l&15) ----
    {
        const int hi   = lane >> 4;
        const int col0 = (wave * 2) * 16 + (lane & 15);
        const int col1 = col0 + 16;
        const float bb0 = bias[col0];
        const float bb1 = bias[col1];
        #pragma unroll
        for (int r = 0; r < 8; ++r) {
            const int m = r + 8 * hi;
            Hs[m * 256 + col0] = c0[r] + bb0;
            Hs[m * 256 + col1] = c1[r] + bb1;
        }
    }
    __syncthreads();

    // ---- Epilogue: optional LayerNorm + exact GELU; fp32 store ----
    const int r  = tid >> 4;
    const int cs = (tid & 15) << 4;
    float* outr = OUT + (rowbase + r) * DIM + cs;
    if (LN_GELU) {
        float vals[16];
        float s = 0.f, ss = 0.f;
        #pragma unroll
        for (int j = 0; j < 16; ++j) {
            float v = Hs[r * 256 + cs + j];
            vals[j] = v; s += v; ss += v * v;
        }
        RS[r][tid & 15]  = s;
        RSS[r][tid & 15] = ss;
        __syncthreads();
        if ((tid & 15) == 0) {
            float S = 0.f, SS = 0.f;
            #pragma unroll
            for (int i = 0; i < 16; ++i) { S += RS[r][i]; SS += RSS[r][i]; }
            const float mu  = S * (1.0f / 256.0f);
            const float var = SS * (1.0f / 256.0f) - mu * mu;
            RS[r][0]  = mu;
            RSS[r][0] = rsqrtf(var + LN_EPS);
        }
        __syncthreads();
        const float mu = RS[r][0], rstd = RSS[r][0];
        #pragma unroll
        for (int j = 0; j < 16; ++j) {
            const int col = cs + j;
            const float v = (vals[j] - mu) * rstd * gamma[col] + beta[col];
            outr[j] = gelu_exact(v);
        }
    } else {
        #pragma unroll
        for (int j = 0; j < 16; ++j) outr[j] = Hs[r * 256 + cs + j];
    }
}

extern "C" void kernel_launch(void* const* d_in, const int* in_sizes, int n_in,
                              void* d_out, int out_size, void* d_ws, size_t ws_size,
                              hipStream_t stream) {
    const float* x0  = (const float*)d_in[0];
    const int*   src = (const int*)d_in[1];
    const int*   dst = (const int*)d_in[2];
    const float* ew  = (const float*)d_in[3];
    const float* W1  = (const float*)d_in[4];
    const float* b1  = (const float*)d_in[5];
    const float* g1  = (const float*)d_in[6];
    const float* be1 = (const float*)d_in[7];
    const float* W2  = (const float*)d_in[8];
    const float* b2  = (const float*)d_in[9];
    const float* g2  = (const float*)d_in[10];
    const float* be2 = (const float*)d_in[11];
    const float* Wpm = (const float*)d_in[12];
    const float* bp  = (const float*)d_in[13];

    float* out = (float*)d_out;

    const size_t NF = (size_t)NNODES * DIM;
    float*  agg = (float*)d_ws;            // 51.2 MB
    float*  x1  = agg + NF;                // 51.2 MB
    __bf16* w1p = (__bf16*)(x1 + NF);      // 128 KB each
    __bf16* w2p = w1p + 256 * 256;
    __bf16* wpp = w2p + 256 * 256;

    // Pre-swizzle weights into bf16 WMMA B-fragment layout (every call; deterministic).
    k_pack_w<<<256, 256, 0, stream>>>(W1,  w1p);
    k_pack_w<<<256, 256, 0, stream>>>(W2,  w2p);
    k_pack_w<<<256, 256, 0, stream>>>(Wpm, wpp);

    const int spmm_blocks = NEDGES / 8;    // one wave32 per edge, 8 waves/block
    const int gemm_blocks = NNODES / 16;   // 3125

    // Layer 1: agg = A @ x0 ; x1 = gelu(LN((x0+agg)@W1 + b1))
    hipMemsetAsync(agg, 0, NF * sizeof(float), stream);
    k_spmm<<<spmm_blocks, 256, 0, stream>>>(src, dst, ew, x0, agg);
    k_gemm_fused<true, true><<<gemm_blocks, 256, 0, stream>>>(x0, agg, w1p, b1, g1, be1, x1);

    // Layer 2: agg = A @ x1 ; out = gelu(LN((x1+agg)@W2 + b2))
    hipMemsetAsync(agg, 0, NF * sizeof(float), stream);
    k_spmm<<<spmm_blocks, 256, 0, stream>>>(src, dst, ew, x1, agg);
    k_gemm_fused<true, true><<<gemm_blocks, 256, 0, stream>>>(x1, agg, w2p, b2, g2, be2, out);

    // Final projection, in place on d_out (each block reads/writes only its own 16 rows).
    k_gemm_fused<false, false><<<gemm_blocks, 256, 0, stream>>>(out, nullptr, wpp, bp, nullptr, nullptr, out);
}
